// LocalGraphAttention_44057774522827
// MI455X (gfx1250) — matrix operand order repaired and checked by reading, not compiled, not run on hardware
//
#include <hip/hip_runtime.h>
#include <hip/hip_bf16.h>

#define D_MODEL 128
#define N_HEADS 8
#define HEAD_DIM 16

typedef __attribute__((ext_vector_type(16))) __bf16 v16bf;
typedef __attribute__((ext_vector_type(8)))  float  v8f;

// ---------------------------------------------------------------------------
// fp32 -> bf16 weight conversion (tiny, one-shot per call)
// ---------------------------------------------------------------------------
__global__ void cvt_f32_to_bf16(const float* __restrict__ src,
                                __bf16* __restrict__ dst, int n) {
    int i = blockIdx.x * blockDim.x + threadIdx.x;
    if (i < n) dst[i] = (__bf16)src[i];
}

// ---------------------------------------------------------------------------
// C[M x Nout] = A[M x K] (fp32, row-major) * W[Nout x K]^T (bf16, row-major)
// One wave computes one 16x16 C tile via v_wmma_f32_16x16x32_bf16.
// blockDim.x = 128 (4 waves); wave w handles n-tile (blockIdx.y*4 + w).
// Requires M % 16 == 0, K % 32 == 0, Nout % 64 == 0.
// ---------------------------------------------------------------------------
__global__ void gat_gemm_bf16_wmma(const float* __restrict__ A,
                                   const __bf16* __restrict__ W,
                                   float* __restrict__ C,
                                   int K, int Nout) {
    const int lane = threadIdx.x & 31;
    const int wave = threadIdx.x >> 5;
    const int m0   = blockIdx.x * 16;
    const int n0   = (blockIdx.y * 4 + wave) * 16;
    const int mr   = lane & 15;   // row within A tile (also B column index)
    const int hi   = lane >> 4;   // lane-group select

    const float*  arow = A + (size_t)(m0 + mr) * K;
    // B[k][n] = W[n*K + k]; lane supplies column n = n0+mr, contiguous in k.
    const __bf16* wrow = W + (size_t)(n0 + mr) * K + 16 * hi;

    v8f acc = {};
    for (int k0 = 0; k0 < K; k0 += 32) {
        // A-fragment: element i <-> K = k0 + 16*hi + i (matches B)
        //   lanes 0-15 : K = k0+[0..15]
        //   lanes 16-31: K = k0+16+[0..15]
        const float4* c0 = (const float4*)(arow + k0 + 16 * hi);
        const float4* c1 = (const float4*)(arow + k0 + 16 * hi + 8);
        float4 f0 = c0[0], f1 = c0[1];
        float4 f2 = c1[0], f3 = c1[1];
        v16bf a;
        a[0]  = (__bf16)f0.x; a[1]  = (__bf16)f0.y;
        a[2]  = (__bf16)f0.z; a[3]  = (__bf16)f0.w;
        a[4]  = (__bf16)f1.x; a[5]  = (__bf16)f1.y;
        a[6]  = (__bf16)f1.z; a[7]  = (__bf16)f1.w;
        a[8]  = (__bf16)f2.x; a[9]  = (__bf16)f2.y;
        a[10] = (__bf16)f2.z; a[11] = (__bf16)f2.w;
        a[12] = (__bf16)f3.x; a[13] = (__bf16)f3.y;
        a[14] = (__bf16)f3.z; a[15] = (__bf16)f3.w;

        // B-fragment: element i <-> K = k0 + 16*hi + i (contiguous 32B load)
        v16bf b = *(const v16bf*)(wrow + k0);

        acc = __builtin_amdgcn_wmma_f32_16x16x32_bf16(
            /*neg_a=*/false, a, /*neg_b=*/false, b,
            /*c_mod=*/(short)0, acc, /*reuse_a=*/false, /*reuse_b=*/false);
    }

    // C/D layout: VGPR r, lane l -> M = r + 8*(l>=16), N = l%16
#pragma unroll
    for (int r = 0; r < 8; ++r)
        C[(size_t)(m0 + r + 8 * hi) * Nout + (n0 + mr)] = acc[r];
}

// ---------------------------------------------------------------------------
// Edge pass 1: one thread per (edge, head).
// attn = leakyrelu(dot(Q[row], K[col]) / 4), clip, exp; atomic denom.
// qkv row layout: [Q(128) | K(128) | V(128)] (row stride 384 floats).
// ---------------------------------------------------------------------------
__global__ void edge_logits(const float* __restrict__ qkv,
                            const int* __restrict__ row,
                            const int* __restrict__ col,
                            float* __restrict__ exp_s,
                            float* __restrict__ denom, int EH) {
    int idx = blockIdx.x * blockDim.x + threadIdx.x;
    if (idx >= EH) return;
    int e = idx >> 3;
    int h = idx & 7;
    int r = row[e];
    int c = col[e];

    const float4* q = (const float4*)(qkv + (size_t)r * (3 * D_MODEL) + h * HEAD_DIM);
    const float4* k = (const float4*)(qkv + (size_t)c * (3 * D_MODEL) + D_MODEL + h * HEAD_DIM);

    float acc = 0.0f;
#pragma unroll
    for (int i = 0; i < 4; ++i) {
        float4 qv = q[i], kv = k[i];
        acc += qv.x * kv.x + qv.y * kv.y + qv.z * kv.z + qv.w * kv.w;
    }
    acc *= 0.25f;                              // 1/sqrt(16)
    acc = (acc >= 0.0f) ? acc : 0.2f * acc;    // LeakyReLU(0.2)
    acc = fminf(fmaxf(acc, -20.0f), 20.0f);
    float ex = __expf(acc);
    exp_s[idx] = ex;
    atomicAdd(&denom[(size_t)r * N_HEADS + h], ex);
}

// ---------------------------------------------------------------------------
// Edge pass 2: w = exp/denom; agg[row,h,:] += w * V[col,h,:]  (fp32 atomics)
// ---------------------------------------------------------------------------
__global__ void edge_aggregate(const float* __restrict__ qkv,
                               const int* __restrict__ row,
                               const int* __restrict__ col,
                               const float* __restrict__ exp_s,
                               const float* __restrict__ denom,
                               float* __restrict__ agg, int EH) {
    int idx = blockIdx.x * blockDim.x + threadIdx.x;
    if (idx >= EH) return;
    int e = idx >> 3;
    int h = idx & 7;
    int r = row[e];
    int c = col[e];

    float w = exp_s[idx] / (denom[(size_t)r * N_HEADS + h] + 1e-10f);

    const float* v  = qkv + (size_t)c * (3 * D_MODEL) + 2 * D_MODEL + h * HEAD_DIM;
    float*      dst = agg + (size_t)r * D_MODEL + h * HEAD_DIM;
#pragma unroll
    for (int d = 0; d < HEAD_DIM; ++d)
        atomicAdd(dst + d, w * v[d]);
}

// ---------------------------------------------------------------------------
// Launch
// inputs: embeds(N*128 f32), row(E int), col(E int),
//         W_qkv(384*128 f32), W_out(128*128 f32); out: N*128 f32
// ---------------------------------------------------------------------------
extern "C" void kernel_launch(void* const* d_in, const int* in_sizes, int n_in,
                              void* d_out, int out_size, void* d_ws, size_t ws_size,
                              hipStream_t stream) {
    const float* embeds = (const float*)d_in[0];
    const int*   row    = (const int*)d_in[1];
    const int*   col    = (const int*)d_in[2];
    const float* Wqkv   = (const float*)d_in[3];
    const float* Wout   = (const float*)d_in[4];
    float*       out    = (float*)d_out;

    const int N = in_sizes[0] / D_MODEL;   // 50000 (multiple of 16)
    const int E = in_sizes[1];             // 1000000
    const int EH = E * N_HEADS;

    // Workspace layout (all region sizes are multiples of 256B here)
    char* ws = (char*)d_ws;
    size_t o = 0;
    float* qkv   = (float*)(ws + o); o += (size_t)N * 3 * D_MODEL * sizeof(float);
    o = (o + 255) & ~(size_t)255;
    float* agg   = (float*)(ws + o); o += (size_t)N * D_MODEL * sizeof(float);
    o = (o + 255) & ~(size_t)255;
    float* denom = (float*)(ws + o); o += (size_t)N * N_HEADS * sizeof(float);
    o = (o + 255) & ~(size_t)255;
    float* exps  = (float*)(ws + o); o += (size_t)EH * sizeof(float);
    o = (o + 255) & ~(size_t)255;
    __bf16* wqkv_bf = (__bf16*)(ws + o); o += (size_t)3 * D_MODEL * D_MODEL * sizeof(__bf16);
    o = (o + 255) & ~(size_t)255;
    __bf16* wout_bf = (__bf16*)(ws + o);

    // Zero accumulation buffers (graph-capture-legal)
    hipMemsetAsync(denom, 0, (size_t)N * N_HEADS * sizeof(float), stream);
    hipMemsetAsync(agg,   0, (size_t)N * D_MODEL * sizeof(float), stream);

    // Weight conversion fp32 -> bf16
    {
        int n1 = 3 * D_MODEL * D_MODEL;
        int n2 = D_MODEL * D_MODEL;
        cvt_f32_to_bf16<<<(n1 + 255) / 256, 256, 0, stream>>>(Wqkv, wqkv_bf, n1);
        cvt_f32_to_bf16<<<(n2 + 255) / 256, 256, 0, stream>>>(Wout, wout_bf, n2);
    }

    // QKV projection: (N x 128) x (128 x 384) -> qkv
    {
        dim3 grid(N / 16, (3 * D_MODEL) / 64);   // 3125 x 6, 4 waves/block
        gat_gemm_bf16_wmma<<<grid, 128, 0, stream>>>(embeds, wqkv_bf, qkv,
                                                     D_MODEL, 3 * D_MODEL);
    }

    // Edge passes
    edge_logits<<<(EH + 255) / 256, 256, 0, stream>>>(qkv, row, col, exps, denom, EH);
    edge_aggregate<<<(EH + 255) / 256, 256, 0, stream>>>(qkv, row, col, exps, denom,
                                                         agg, EH);

    // Output projection: (N x 128) x (128 x 128) -> out
    {
        dim3 grid(N / 16, D_MODEL / 64);         // 3125 x 2
        gat_gemm_bf16_wmma<<<grid, 128, 0, stream>>>(agg, wout_bf, out,
                                                     D_MODEL, D_MODEL);
    }
}
